// ZSSR_INV_1769526526714
// MI455X (gfx1250) — matrix operand order repaired and checked by reading, not compiled, not run on hardware
//
#include <hip/hip_runtime.h>

// ---------------------------------------------------------------------------
// MI455X (gfx1250) involution network.
//   - all GEMM-shaped math on v_wmma_f32_16x16x32_bf16
//   - activations live in global as bf16, pixel-major [y][x][ch]  (the GEMMs
//     consume bf16 anyway, so this costs no extra precision and halves traffic)
//   - LDS staging uses GLOBAL_LOAD_ASYNC_TO_LDS_B128 (ASYNCcnt) -- pure DMA
//     copy, OOB lanes redirected to a zero block so padding comes for free
// ---------------------------------------------------------------------------

typedef __attribute__((ext_vector_type(16))) __bf16 v16bf;
typedef __attribute__((ext_vector_type(8)))  float  v8f;

#define DEVI static __device__ __forceinline__

// problem constants
constexpr int   C   = 64;
constexpr int   CR  = 16;
constexpr int   G   = 4;
constexpr int   GC  = 16;            // channels per group
constexpr int   KS  = 7;
constexpr int   KK  = 49;
constexpr int   NL  = 6;
constexpr int   H   = 126, W = 126, HW = H * W;
constexpr int   OH  = 124, OW = 124;
constexpr float EPS = 1e-5f;

// tiling
constexpr int TS   = 16;             // 16x16 pixel tile per block
constexpr int TPX  = TS * TS;        // 256 pixels
constexpr int HT   = TS + KS - 1;    // 22 halo extent
constexpr int XSTR = 72;             // padded LDS channel stride (bank spread, 16B align)
constexpr int TSTR = 40;             // padded t-channel stride (32 logical: 16 real + 16 zero)

DEVI v16bf load16(const __bf16* p) {           // 32B = two aligned b128 loads
  union { uint4 u[2]; v16bf v; } r;
  r.u[0] = *(const uint4*)(p);
  r.u[1] = *(const uint4*)(p + 8);
  return r.v;
}

DEVI v8f wmma_bf16(v16bf a, v16bf b, v8f c) {
  return __builtin_amdgcn_wmma_f32_16x16x32_bf16(false, a, false, b, (short)0, c,
                                                 false, false);
}

// async 16B global -> LDS copy (ASYNCcnt). lds: generic pointer to __shared__
// (low 32 bits == LDS byte offset); gaddr: 64-bit global address.
DEVI void async_g2l_b128(const void* lds, const void* gaddr) {
  asm volatile("global_load_async_to_lds_b128 %0, %1, off"
               :: "v"((unsigned)(size_t)lds), "v"((unsigned long long)(size_t)gaddr)
               : "memory");
}
DEVI void wait_async0() { asm volatile("s_wait_asynccnt 0x0" ::: "memory"); }

// A-matrix (16x32 bf16) per-lane K mapping: contiguous order, lane half h
// holds K = 16h + e (matches the contiguous B loads).
DEVI int amap_k(int lane, int e) {
  return (lane >> 4) * 16 + e;
}

// ---------------------------------------------------------------------------
__global__ void prep_frags(const float* __restrict__ wr, const float* __restrict__ ws,
                           const float* __restrict__ wout,
                           __bf16* __restrict__ wrF, __bf16* __restrict__ wsF,
                           __bf16* __restrict__ woutF, unsigned* __restrict__ zeros) {
  int tid = blockIdx.x * blockDim.x + threadIdx.x;
  int stride = gridDim.x * blockDim.x;

  if (tid < 64) zeros[tid] = 0u;   // 256B zero block for OOB async sources

  // wrF: [l][kc(2)][lane(32)][e(16)]  A = Wr(16x64)
  const int NWR = NL * 2 * 32 * 16;
  for (int i = tid; i < NWR; i += stride) {
    int e = i & 15, lane = (i >> 4) & 31, kc = (i >> 9) & 1, l = i >> 10;
    int m = lane & 15;
    int kg = kc * 32 + amap_k(lane, e);
    wrF[i] = (__bf16)wr[(l * CR + m) * C + kg];
  }
  // wsF: [l][g(4)][mt(4)][lane][e]  A = Ws rows padded per-group to 64, K 16->32 pad
  const int NWS = NL * G * 4 * 32 * 16;
  for (int i = tid; i < NWS; i += stride) {
    int e = i & 15, lane = (i >> 4) & 31, mt = (i >> 9) & 3, g = (i >> 11) & 3,
        l = i >> 13;
    int k = amap_k(lane, e);
    int j = mt * 16 + (lane & 15);
    float v = 0.f;
    if (k < CR && j < KK) v = ws[(l * (G * KK) + g * KK + j) * CR + k];
    wsF[i] = (__bf16)v;
  }
  // woutF: [kk(18)][mw(8)][lane][e]  A = Wout(128x576), K order = (ky*3+kx)*64+ci
  const int NWO = 18 * 8 * 32 * 16;
  for (int i = tid; i < NWO; i += stride) {
    int e = i & 15, lane = (i >> 4) & 31, mw = (i >> 9) & 7, kk = i >> 12;
    int m = mw * 16 + (lane & 15);
    int kg = kk * 32 + amap_k(lane, e);
    int pos = kg >> 6, ci = kg & 63;
    int ky = pos / 3, kx = pos % 3;
    woutF[i] = (__bf16)wout[((m * C + ci) * 3 + ky) * 3 + kx];
  }
}

// ---------------------------------------------------------------------------
// 3x3 conv 3->64, one thread per pixel, writes bf16 [y][x][ch] layout.
__global__ __launch_bounds__(256) void conv_in_k(const float* __restrict__ img,
                                                 const float* __restrict__ w,
                                                 const float* __restrict__ b,
                                                 __bf16* __restrict__ act) {
  __shared__ float wl[C * 27];
  __shared__ float bl[C];
  const int tid = threadIdx.x;
  for (int i = tid; i < C * 27; i += 256) wl[i] = w[i];
  if (tid < C) bl[tid] = b[tid];
  __syncthreads();

  int p = blockIdx.x * blockDim.x + tid;
  if (p >= HW) return;
  int oy = p / W, ox = p - oy * W;

  float in[27];
#pragma unroll
  for (int ci = 0; ci < 3; ++ci)
#pragma unroll
    for (int ky = 0; ky < 3; ++ky)
#pragma unroll
      for (int kx = 0; kx < 3; ++kx)
        in[ci * 9 + ky * 3 + kx] = img[(ci * 128 + oy + ky) * 128 + ox + kx];

  union { __bf16 h[C]; uint4 q[C / 8]; } o;
#pragma unroll 4
  for (int oc = 0; oc < C; ++oc) {
    float acc = bl[oc];
#pragma unroll
    for (int t = 0; t < 27; ++t) acc = fmaf(wl[oc * 27 + t], in[t], acc);
    o.h[oc] = (__bf16)acc;
  }
  uint4* dp = (uint4*)(act + (size_t)p * C);
#pragma unroll
  for (int q = 0; q < C / 8; ++q) dp[q] = o.q[q];
}

// ---------------------------------------------------------------------------
__global__ __launch_bounds__(256) void involution_k(
    const __bf16* __restrict__ src, __bf16* __restrict__ dst,
    const __bf16* __restrict__ wrF, const __bf16* __restrict__ wsF,
    const float* __restrict__ br, const float* __restrict__ gam,
    const float* __restrict__ bet, const float* __restrict__ mu,
    const float* __restrict__ var, const float* __restrict__ bs,
    const void* __restrict__ zeros, int layer) {
  __shared__ alignas(32) __bf16 xs[HT * HT * XSTR];  // halo tile, [pix][ch] 69.7KB
  __shared__ alignas(32) __bf16 ts[TPX * TSTR];      // t (span B), [pix][k]  20.5KB
  __shared__ float kw[64 * TPX];                     // per-group kernels      64KB
  __shared__ float sc[CR], sh[CR];

  const int tid  = threadIdx.x;
  const int lane = tid & 31, wave = tid >> 5;
  const int x0 = (blockIdx.x & 7) * TS, y0 = (blockIdx.x >> 3) * TS;

  // ---- phase 0: fold BN, zero t pad, async-DMA halo tile into LDS ----
  if (tid < CR) {
    float s = gam[layer * CR + tid] * rsqrtf(var[layer * CR + tid] + EPS);
    sc[tid] = s;
    sh[tid] = bet[layer * CR + tid] + (br[layer * CR + tid] - mu[layer * CR + tid]) * s;
  }
  for (int i = tid; i < TPX * TSTR; i += 256) ts[i] = (__bf16)0.f;

  // 484 halo pixels x 8 chunks of 16B; OOB lanes copy from the zero block.
  for (int i = tid; i < HT * HT * 8; i += 256) {
    int chunk = i & 7, pp = i >> 3;
    int yy = pp / HT, xx = pp - yy * HT;
    int gy = y0 + yy - 3, gx = x0 + xx - 3;
    bool inb = (gy >= 0) && (gy < H) && (gx >= 0) && (gx < W);
    const void* ga = inb ? (const void*)(src + ((size_t)(gy * W + gx) * C + chunk * 8))
                         : zeros;
    async_g2l_b128(xs + pp * XSTR + chunk * 8, ga);
  }
  wait_async0();
  __syncthreads();

  // ---- phase 1: reduce GEMM  t(16 x 256px) = Wr(16x64) * X(64x256px) ----
  const int n = lane & 15, lh = lane >> 4;
  for (int j = 0; j < 2; ++j) {
    int nt = wave * 2 + j;
    int p  = nt * 16 + n;
    int hp = ((p >> 4) + 3) * HT + (p & 15) + 3;
    v8f acc = {};
#pragma unroll
    for (int kc = 0; kc < 2; ++kc) {
      v16bf a = load16(wrF + ((layer * 2 + kc) * 32 + lane) * 16);
      v16bf b = load16(xs + hp * XSTR + kc * 32 + lh * 16);
      acc = wmma_bf16(a, b, acc);
    }
#pragma unroll
    for (int v = 0; v < 8; ++v) {
      int m = v + 8 * lh;
      float t = fmaf(acc[v], sc[m], sh[m]);
      ts[p * TSTR + m] = (__bf16)(t > 0.f ? t : 0.f);
    }
  }
  __syncthreads();

  // ---- phase 2: per group, span GEMM then spatially-varying aggregation ----
  const int ty = tid >> 4, tx = tid & 15;
  const int oy = y0 + ty, ox = x0 + tx;
  const bool valid = (oy < H) && (ox < W);

  for (int g = 0; g < G; ++g) {
    // span: kw(64pad x 256px) = WsPad(64x32) * t(32x256px); 64 tile-jobs / 8 waves
    for (int j = wave; j < 64; j += 8) {
      int mt = j >> 4, nt = j & 15;
      int p = nt * 16 + n;
      v16bf a = load16(wsF + (((layer * G + g) * 4 + mt) * 32 + lane) * 16);
      v16bf b = load16(ts + p * TSTR + lh * 16);
      v8f acc = {};
      acc = wmma_bf16(a, b, acc);
#pragma unroll
      for (int v = 0; v < 8; ++v) {
        int jl = mt * 16 + v + 8 * lh;
        float bias = (jl < KK) ? bs[(layer * G + g) * KK + jl] : 0.f;
        kw[jl * TPX + p] = acc[v] + bias;
      }
    }
    __syncthreads();

    // aggregate: out[c,p] = sum_k kw[k,p] * x[c, p+off(k)], ReLU, bf16 pack+store
    float accv[GC];
#pragma unroll
    for (int c = 0; c < GC; ++c) accv[c] = 0.f;
    for (int ky = 0; ky < KS; ++ky)
#pragma unroll
      for (int kx = 0; kx < KS; ++kx) {
        float wv = kw[(ky * KS + kx) * TPX + tid];
        v16bf xv = load16(xs + ((ty + ky) * HT + tx + kx) * XSTR + g * GC);
#pragma unroll
        for (int c = 0; c < GC; ++c) accv[c] = fmaf(wv, (float)xv[c], accv[c]);
      }
    if (valid) {
      union { __bf16 h[GC]; uint4 q[2]; } o;
#pragma unroll
      for (int c = 0; c < GC; ++c) o.h[c] = (__bf16)(accv[c] > 0.f ? accv[c] : 0.f);
      uint4* dp = (uint4*)(dst + (size_t)(oy * W + ox) * C + g * GC);
      dp[0] = o.q[0];
      dp[1] = o.q[1];
    }
    __syncthreads();  // kw reused next group
  }
}

// ---------------------------------------------------------------------------
__global__ __launch_bounds__(256) void conv_out_k(const __bf16* __restrict__ src,
                                                  const __bf16* __restrict__ woF,
                                                  const float* __restrict__ b,
                                                  const void* __restrict__ zeros,
                                                  float* __restrict__ out) {
  __shared__ alignas(32) __bf16 xs[3 * 18 * XSTR];  // [ky][x][ch] 7.8KB
  const int tid = threadIdx.x, lane = tid & 31, wave = tid >> 5;
  const int oy = blockIdx.x >> 3;
  const int x0 = (blockIdx.x & 7) * 16;

  // async-DMA 3x18 pixel strip (64ch each) into LDS; x-OOB from zero block
  for (int i = tid; i < 54 * 8; i += 256) {
    int chunk = i & 7, r = i >> 3;
    int ky = r / 18, xx = r - ky * 18;
    int gx = x0 + xx;
    const void* ga = (gx < W)
        ? (const void*)(src + ((size_t)((oy + ky) * W + gx) * C + chunk * 8))
        : zeros;
    async_g2l_b128(xs + (ky * 18 + xx) * XSTR + chunk * 8, ga);
  }
  wait_async0();
  __syncthreads();

  const int n = lane & 15, lh = lane >> 4;
  v8f acc = {};
  for (int kk = 0; kk < 18; ++kk) {          // K = 576 = 18 * 32
    v16bf a = load16(woF + ((kk * 8 + wave) * 32 + lane) * 16);
    int pos = kk >> 1;
    int ky = pos / 3, kx = pos % 3;
    int ci0 = (kk & 1) * 32 + lh * 16;
    v16bf bb = load16(xs + (ky * 18 + n + kx) * XSTR + ci0);
    acc = wmma_bf16(a, bb, acc);
  }
  int ox = x0 + n;
  if (ox < OW) {
#pragma unroll
    for (int v = 0; v < 8; ++v) {
      int m = wave * 16 + v + 8 * lh;
      out[m * (OH * OW) + oy * OW + ox] = acc[v] + b[m];
    }
  }
}

// ---------------------------------------------------------------------------
extern "C" void kernel_launch(void* const* d_in, const int* in_sizes, int n_in,
                              void* d_out, int out_size, void* d_ws, size_t ws_size,
                              hipStream_t stream) {
  const float* img = (const float*)d_in[0];
  const float* wci = (const float*)d_in[1];
  const float* bci = (const float*)d_in[2];
  const float* wr  = (const float*)d_in[3];
  const float* br  = (const float*)d_in[4];
  const float* gam = (const float*)d_in[5];
  const float* bet = (const float*)d_in[6];
  const float* mu  = (const float*)d_in[7];
  const float* var = (const float*)d_in[8];
  const float* wsp = (const float*)d_in[9];
  const float* bsp = (const float*)d_in[10];
  const float* wco = (const float*)d_in[11];
  const float* bco = (const float*)d_in[12];
  float* out = (float*)d_out;

  char* wsb = (char*)d_ws;
  __bf16*   actA  = (__bf16*)(wsb);                       // 126*126*64 bf16 ~ 2MB
  __bf16*   actB  = (__bf16*)(wsb + (size_t)(2 << 20));
  __bf16*   wrF   = (__bf16*)(wsb + (size_t)(4 << 20));   // fragment tables ~ 252KB
  __bf16*   wsF   = wrF + NL * 2 * 32 * 16;
  __bf16*   woF   = wsF + NL * G * 4 * 32 * 16;
  unsigned* zeros = (unsigned*)(wsb + (size_t)(5 << 20)); // 256B OOB source

  prep_frags<<<64, 256, 0, stream>>>(wr, wsp, wco, wrF, wsF, woF, zeros);
  conv_in_k<<<(HW + 255) / 256, 256, 0, stream>>>(img, wci, bci, actA);

  __bf16* a = actA; __bf16* bb = actB;
  for (int l = 0; l < NL; ++l) {
    involution_k<<<64, 256, 0, stream>>>(a, bb, wrF, wsF, br, gam, bet, mu, var,
                                         bsp, zeros, l);
    __bf16* t = a; a = bb; bb = t;
  }
  conv_out_k<<<OH * 8, 256, 0, stream>>>(a, woF, bco, zeros, out);
}
